// WindowAttention_90975997264484
// MI455X (gfx1250) — matrix operand order repaired and checked by reading, not compiled, not run on hardware
//
#include <hip/hip_runtime.h>
#include <hip/hip_bf16.h>

typedef _Float16 h16;
typedef __attribute__((ext_vector_type(16))) _Float16 v16h;
typedef __attribute__((ext_vector_type(8)))  float    v8f;

#define DIM    128
#define HEADS  4
#define N_TOK  49
#define NPAD   64
#define HD     32
#define NWIN   16
#define QK_LD  256                  // row stride of q|k buffer (halves)
#define QKVW_ELEMS (3 * DIM * DIM)  // 49152
#define PROJW_ELEMS (DIM * DIM)     // 16384

// ---- fragment loaders (CDNA5 wave32 WMMA f16 16x16x32 layouts) ----

// A matrix 16x32 f16: lane%16 = M; lane group g: K in {k0+8g..k0+8g+7} u {k0+16+8g..k0+23+8g}.
__device__ __forceinline__ v16h fragA(const h16* rowbase, int k0, int g) {
  const h16* p = rowbase + k0 + g * 8;
  v16h a;
#pragma unroll
  for (int i = 0; i < 8; ++i) { a[i] = p[i]; a[i + 8] = p[i + 16]; }
  return a;
}

// B matrix 32x16 f16: lane%16 = N column; lane group g: K in {k0+8g..k0+8g+7} u {k0+16+8g..k0+23+8g}.
// Works for LDS or global f16 sources.
__device__ __forceinline__ v16h fragB(const h16* colbase, int k0, int g) {
  const h16* p = colbase + k0 + g * 8;
  v16h b;
#pragma unroll
  for (int i = 0; i < 8; ++i) { b[i] = p[i]; b[i + 8] = p[i + 16]; }
  return b;
}

__device__ __forceinline__ v8f wmma_f16(v16h a, v16h b, v8f c) {
  return __builtin_amdgcn_wmma_f32_16x16x32_f16(false, a, false, b, (short)0, c, false, false);
}

// ---- prologue: one-time f32 -> f16 conversion of the (L2-resident) weights ----
__global__ void convert_weights_kernel(const float* __restrict__ qkv_w,
                                       const float* __restrict__ proj_w,
                                       h16* __restrict__ wgt) {
  int i = blockIdx.x * 256 + threadIdx.x;
  if (i < QKVW_ELEMS) {
    wgt[i] = (h16)qkv_w[i];
  } else if (i < QKVW_ELEMS + PROJW_ELEMS) {
    wgt[i] = (h16)proj_w[i - QKVW_ELEMS];
  }
}

__global__ __launch_bounds__(256, 2)
void swin_window_attn_kernel(const float* __restrict__ x,
                             const float* __restrict__ mask,
                             const h16*   __restrict__ wgt,      // f16 qkv_w | proj_w
                             const float* __restrict__ qkv_b,
                             const float* __restrict__ proj_b,
                             const float* __restrict__ bias_table,
                             const int*   __restrict__ rel_index,
                             float* __restrict__ out)
{
  extern __shared__ char smem_raw[];
  h16*   xs   = (h16*)smem_raw;                    // [64][128] f16: x panel, later O panel (16 KB)
  h16*   qks  = xs + NPAD * DIM;                   // [64][256] f16: q | k                  (32 KB)
  h16*   vT   = qks + NPAD * QK_LD;                // [4*32][64] f16: v transposed          (16 KB)
  float* Ssc  = (float*)(vT + HEADS * HD * NPAD);  // [4][64][64] f32 scores                (64 KB)
  h16*   Ps   = (h16*)(Ssc + HEADS * NPAD * NPAD); // [4][64][64] f16 softmax probs         (32 KB)
  float* stagef = Ssc;                             // x f32 staging aliases scores (25 KB <= 64 KB)

  const int b    = blockIdx.x;
  const int tid  = threadIdx.x;
  const int lane = tid & 31;
  // wave index is wave-uniform by construction; readfirstlane puts it (and all
  // tile indices derived from it) into SGPRs -> scalar branches, SALU loop control.
  const int wave = __builtin_amdgcn_readfirstlane(tid >> 5);
  const int l    = lane & 15;   // M row (A) / N col (B,D) within tile
  const int g    = lane >> 4;   // lane half-group
  const int wm   = b & (NWIN - 1);

  const h16* qkv_w16  = wgt;
  const h16* proj_w16 = wgt + QKVW_ELEMS;

  // ---------- async-copy x (f32, 49x128 = 1568 x 16B) into LDS staging ----------
  {
    const float* xb = x + (size_t)b * N_TOK * DIM;
    unsigned stage_base = (unsigned)(size_t)(void*)stagef;   // low 32 bits = LDS byte offset
    for (int idx = tid; idx < (N_TOK * DIM) / 4; idx += 256) {
      unsigned lds_addr = stage_base + (unsigned)idx * 16u;
      unsigned long long ga = (unsigned long long)(const void*)xb + (unsigned long long)idx * 16ull;
      asm volatile("global_load_async_to_lds_b128 %0, %1, off"
                   :: "v"(lds_addr), "v"(ga) : "memory");
    }
    asm volatile("s_wait_asynccnt 0" ::: "memory");
  }
  __syncthreads();

  // ---------- convert staged f32 -> f16 A-panel, zero-pad rows 49..63 ----------
  for (int idx = tid; idx < NPAD * DIM; idx += 256) {
    int row = idx >> 7;
    xs[idx] = (row < N_TOK) ? (h16)stagef[idx] : (h16)0.f;
  }
  __syncthreads();

  // ---------- QKV GEMM: [64x128] @ [128x384] + bias ----------
  // Wave owns M-tile (wave&3); A-fragments hoisted once; three uniform nt loops
  // (q: scaled, k: plain, v: transposed store) -> zero divergence.
  {
    const int mt  = wave & 3;
    const int nt0 = wave >> 2;
    const h16* arow = xs + (mt * 16 + l) * DIM;
    v16h af[4];
#pragma unroll
    for (int kk = 0; kk < 4; ++kk) af[kk] = fragA(arow, kk * 32, g);

    // q tiles: nt in [0,8), scale by 1/sqrt(32)
    for (int nt = nt0; nt < 8; nt += 2) {
      int ocol = nt * 16 + l;
      const h16* wrw = qkv_w16 + (size_t)ocol * DIM;
      v8f c = {};
#pragma unroll
      for (int kk = 0; kk < 4; ++kk)
        c = wmma_f16(af[kk], fragB(wrw, kk * 32, g), c);
      float bias = qkv_b[ocol];
#pragma unroll
      for (int r = 0; r < 8; ++r)
        qks[(mt * 16 + r + g * 8) * QK_LD + ocol] =
            (h16)((c[r] + bias) * 0.17677669529663688f);
    }
    // k tiles: nt in [8,16)
    for (int nt = 8 + nt0; nt < 16; nt += 2) {
      int ocol = nt * 16 + l;
      const h16* wrw = qkv_w16 + (size_t)ocol * DIM;
      v8f c = {};
#pragma unroll
      for (int kk = 0; kk < 4; ++kk)
        c = wmma_f16(af[kk], fragB(wrw, kk * 32, g), c);
      float bias = qkv_b[ocol];
#pragma unroll
      for (int r = 0; r < 8; ++r)
        qks[(mt * 16 + r + g * 8) * QK_LD + ocol] = (h16)(c[r] + bias);
    }
    // v tiles: nt in [16,24), store transposed (contiguous b128 per lane)
    for (int nt = 16 + nt0; nt < 24; nt += 2) {
      int ocol = nt * 16 + l;
      const h16* wrw = qkv_w16 + (size_t)ocol * DIM;
      v8f c = {};
#pragma unroll
      for (int kk = 0; kk < 4; ++kk)
        c = wmma_f16(af[kk], fragB(wrw, kk * 32, g), c);
      float bias = qkv_b[ocol];
#pragma unroll
      for (int r = 0; r < 8; ++r)
        vT[(ocol - 2 * DIM) * NPAD + mt * 16 + r + g * 8] = (h16)(c[r] + bias);
    }
  }
  __syncthreads();

  // ---------- scores: S = q k^T + rel_bias + mask ----------
  for (int cc = wave; cc < 16; cc += 8) {        // (head, M-tile) combos; A hoisted
    int hh = cc >> 2, mt = cc & 3;
    v16h aq = fragA(qks + (mt * 16 + l) * QK_LD + hh * HD, 0, g);
#pragma unroll
    for (int nt = 0; nt < 4; ++nt) {
      const h16* bcol = qks + (nt * 16 + l) * QK_LD + DIM + hh * HD;  // k row = B column
      v8f c = {};
      c = wmma_f16(aq, fragB(bcol, 0, g), c);
      int n = nt * 16 + l;
#pragma unroll
      for (int r = 0; r < 8; ++r) {
        int m = mt * 16 + r + g * 8;
        float val = c[r];
        if (m < N_TOK && n < N_TOK) {
          int ri = rel_index[m * N_TOK + n];
          val += bias_table[ri * HEADS + hh] + mask[((size_t)wm * N_TOK + m) * N_TOK + n];
        }
        Ssc[(hh * NPAD + m) * NPAD + n] = val;
      }
    }
  }
  __syncthreads();

  // ---------- softmax: one row (head, query) per thread; pad -> 0 ----------
  {
    int hh = tid >> 6, m = tid & 63;
    float* srow = Ssc + (hh * NPAD + m) * NPAD;
    h16*   prow = Ps  + (hh * NPAD + m) * NPAD;
    if (m < N_TOK) {
      float mx = -1e30f;
      for (int n = 0; n < N_TOK; ++n) mx = fmaxf(mx, srow[n]);
      float s = 0.f;
      for (int n = 0; n < N_TOK; ++n) { float e = __expf(srow[n] - mx); s += e; srow[n] = e; }
      float inv = 1.f / s;
      for (int n = 0; n < N_TOK; ++n) prow[n] = (h16)(srow[n] * inv);
      for (int n = N_TOK; n < NPAD; ++n) prow[n] = (h16)0.f;
    } else {
      for (int n = 0; n < NPAD; ++n) prow[n] = (h16)0.f;
    }
  }
  __syncthreads();

  // ---------- O = P @ V, written into xs as [64][128] (tok, h*32+d) ----------
  for (int cc = wave; cc < 16; cc += 8) {        // (head, M-tile) combos; A hoisted
    int hh = cc >> 2, mt = cc & 3;
    const h16* arow = Ps + (hh * NPAD + mt * 16 + l) * NPAD;
    v16h a0 = fragA(arow, 0, g);
    v16h a1 = fragA(arow, 32, g);
#pragma unroll
    for (int nt = 0; nt < 2; ++nt) {
      const h16* bcol = vT + (hh * HD + nt * 16 + l) * NPAD;  // contiguous along token K
      v8f c = {};
      c = wmma_f16(a0, fragB(bcol, 0, g), c);
      c = wmma_f16(a1, fragB(bcol, 32, g), c);
      int col = hh * HD + nt * 16 + l;
#pragma unroll
      for (int r = 0; r < 8; ++r)
        xs[(mt * 16 + r + g * 8) * DIM + col] = (h16)c[r];
    }
  }
  __syncthreads();

  // ---------- output projection: [64x128] @ [128x128] + bias -> global f32 ----------
  {
    float* outb = out + (size_t)b * N_TOK * DIM;
    const int mt  = wave & 3;
    const int nt0 = wave >> 2;
    const h16* arow = xs + (mt * 16 + l) * DIM;
    v16h af[4];
#pragma unroll
    for (int kk = 0; kk < 4; ++kk) af[kk] = fragA(arow, kk * 32, g);

    for (int nt = nt0; nt < 8; nt += 2) {
      int ocol = nt * 16 + l;
      const h16* wrw = proj_w16 + (size_t)ocol * DIM;
      v8f c = {};
#pragma unroll
      for (int kk = 0; kk < 4; ++kk)
        c = wmma_f16(af[kk], fragB(wrw, kk * 32, g), c);
      float pb = proj_b[ocol];
      if (mt < 3) {                       // scalar branch: rows mt*16+15 <= 47 < 49
#pragma unroll
        for (int r = 0; r < 8; ++r)
          outb[(mt * 16 + r + g * 8) * DIM + ocol] = c[r] + pb;
      } else {                            // last M-tile: per-row guard (rows 48..63)
#pragma unroll
        for (int r = 0; r < 8; ++r) {
          int m = mt * 16 + r + g * 8;
          if (m < N_TOK) outb[m * DIM + ocol] = c[r] + pb;
        }
      }
    }
  }
}

extern "C" void kernel_launch(void* const* d_in, const int* in_sizes, int n_in,
                              void* d_out, int out_size, void* d_ws, size_t ws_size,
                              hipStream_t stream) {
  const float* x          = (const float*)d_in[0];
  const float* mask       = (const float*)d_in[1];
  const float* qkv_w      = (const float*)d_in[2];
  const float* qkv_b      = (const float*)d_in[3];
  const float* proj_w     = (const float*)d_in[4];
  const float* proj_b     = (const float*)d_in[5];
  const float* bias_table = (const float*)d_in[6];
  const int*   rel_index  = (const int*)d_in[7];
  float* out = (float*)d_out;
  h16*   wgt = (h16*)d_ws;   // 131072 B of f16 weights in scratch

  convert_weights_kernel<<<(QKVW_ELEMS + PROJW_ELEMS + 255) / 256, 256, 0, stream>>>(
      qkv_w, proj_w, wgt);

  int B = in_sizes[0] / (N_TOK * DIM);  // 2048
  size_t shmem = (size_t)NPAD * DIM * 2           // xs   16 KB
               + (size_t)NPAD * QK_LD * 2          // q|k  32 KB
               + (size_t)HEADS * HD * NPAD * 2     // vT   16 KB
               + (size_t)HEADS * NPAD * NPAD * 4   // S    64 KB
               + (size_t)HEADS * NPAD * NPAD * 2;  // P    32 KB  -> 160 KB
  swin_window_attn_kernel<<<B, 256, shmem, stream>>>(
      x, mask, wgt, qkv_b, proj_b, bias_table, rel_index, out);
}
